// CrossAttention_31404800868991
// MI455X (gfx1250) — matrix operand order repaired and checked by reading, not compiled, not run on hardware
//
#include <hip/hip_runtime.h>
#include <hip/hip_bf16.h>

typedef __attribute__((ext_vector_type(16))) _Float16 v16h;
typedef __attribute__((ext_vector_type(8)))  _Float16 v8h;
typedef __attribute__((ext_vector_type(8)))  float    v8f;
typedef __attribute__((ext_vector_type(4)))  int      v4i;

#define WMMA_F16(a, b, c) \
  __builtin_amdgcn_wmma_f32_16x16x32_f16(false, (a), false, (b), (short)0, (c), false, false)

#if defined(__AMDGCN__) && __has_builtin(__builtin_amdgcn_global_load_async_to_lds_b128) && \
    __has_builtin(__builtin_amdgcn_s_wait_asynccnt)
#define HAVE_ASYNC_LDS 1
typedef __attribute__((address_space(1))) v4i g_v4i;   // global int4
typedef __attribute__((address_space(3))) v4i l_v4i;   // LDS int4
#else
#define HAVE_ASYNC_LDS 0
#endif

// ---------------- fragment loaders (wave32, 16x16x32 f16 WMMA) ----------------
// A (16x32, MxK): lane<16 -> K {0..7,16..23}; lane>=16 -> K {8..15,24..31}; row = lane&15
__device__ __forceinline__ v16h ldA(const _Float16* base, int stride, int lane) {
  const _Float16* p = base + (lane & 15) * stride + ((lane < 16) ? 0 : 8);
  v8h lo = *(const v8h*)p;
  v8h hi = *(const v8h*)(p + 16);
  return __builtin_shufflevector(lo, hi, 0,1,2,3,4,5,6,7,8,9,10,11,12,13,14,15);
}
// B (32x16, KxN) from N-major storage: lane<16 -> K {0..7,16..23}; lane>=16 -> K {8..15,24..31}; col = lane&15
__device__ __forceinline__ v16h ldB(const _Float16* base, int stride, int lane) {
  const _Float16* p = base + (lane & 15) * stride + ((lane < 16) ? 0 : 8);
  v8h lo = *(const v8h*)p;
  v8h hi = *(const v8h*)(p + 16);
  return __builtin_shufflevector(lo, hi, 0,1,2,3,4,5,6,7,8,9,10,11,12,13,14,15);
}

// ---------------- prep kernels ----------------
__global__ void k_f32_to_f16(const float* __restrict__ src, _Float16* __restrict__ dst, int n) {
  int i = blockIdx.x * blockDim.x + threadIdx.x;
  if (i < n) dst[i] = (_Float16)src[i];
}
// src [K][N] f32 -> dst [N][K] f16
__global__ void k_transpose_f16(const float* __restrict__ src, _Float16* __restrict__ dst,
                                int K, int N) {
  int i = blockIdx.x * blockDim.x + threadIdx.x;
  if (i < K * N) {
    int n = i / K, k = i % K;
    dst[i] = (_Float16)src[(size_t)k * N + n];
  }
}

// ------- 32x64 GEMM tile per wave: C(32x64) = A(32xK) * Bt^T ; B-frags reused 2x -------
__device__ __forceinline__ void gemm32x64(const _Float16* __restrict__ A,   // + m0*K
                                          const _Float16* __restrict__ Bt,  // + n0*K (N-major)
                                          int K, int lane, v8f c[2][4]) {
  for (int k0 = 0; k0 < K; k0 += 32) {
    v16h a0 = ldA(A + k0, K, lane);
    v16h a1 = ldA(A + (size_t)16 * K + k0, K, lane);
#pragma unroll
    for (int f = 0; f < 4; ++f) {
      v16h b = ldB(Bt + (size_t)(f * 16) * K + k0, K, lane);
      c[0][f] = WMMA_F16(a0, b, c[0][f]);
      c[1][f] = WMMA_F16(a1, b, c[1][f]);
    }
  }
}

// ---------------- Q projection + bias + RoPE + scale -> Qh [B][12][512][64] f16 -------------
__global__ void __launch_bounds__(128)
k_gemm_q(const _Float16* __restrict__ lat, const _Float16* __restrict__ WqT,
         const float* __restrict__ bq, const float* __restrict__ ropeq,
         _Float16* __restrict__ Qh) {
  int lane = threadIdx.x & 31, wave = threadIdx.x >> 5;
  int m0 = blockIdx.y * 128 + wave * 32;
  int n0 = blockIdx.x * 64;
  v8f c[2][4] = {};
  gemm32x64(lat + (size_t)m0 * 768, WqT + (size_t)n0 * 768, 768, lane, c);
  int colb = lane & 15;
  int rowOff = (lane < 16) ? 0 : 8;
  int hh = n0 >> 6;  // head (n-tile == one head's 64 dims)
#pragma unroll
  for (int mh = 0; mh < 2; ++mh) {
#pragma unroll
    for (int f = 0; f < 4; ++f) {
      float bb = bq[n0 + f * 16 + colb];
#pragma unroll
      for (int r = 0; r < 8; ++r) c[mh][f][r] += bb;
    }
#pragma unroll
    for (int r = 0; r < 8; ++r) {
      int row = m0 + mh * 16 + rowOff + r;
      int q = row & 511, b = row >> 9;
      const float* rq = ropeq + q * 128;
      float s0 = rq[colb],      s1 = rq[16 + colb], s2 = rq[32 + colb], s3 = rq[48 + colb];
      float c0 = rq[64 + colb], c1 = rq[80 + colb], c2 = rq[96 + colb], c3 = rq[112 + colb];
      float x0 = c[mh][0][r], x1 = c[mh][1][r], x2 = c[mh][2][r], x3 = c[mh][3][r];
      _Float16* o = Qh + (((size_t)(b * 12 + hh) * 512 + q) * 64);
      o[colb]      = (_Float16)((x0 * c0 - x2 * s0) * 0.125f);
      o[16 + colb] = (_Float16)((x1 * c1 - x3 * s1) * 0.125f);
      o[32 + colb] = (_Float16)((x2 * c2 + x0 * s2) * 0.125f);
      o[48 + colb] = (_Float16)((x3 * c3 + x1 * s3) * 0.125f);
    }
  }
}

// -------- KV projection + bias; K gets RoPE -> Kh [B][12][4096][64]; V -> Vt [B][12][64][4096]
__global__ void __launch_bounds__(128)
k_gemm_kv(const _Float16* __restrict__ dat, const _Float16* __restrict__ WkvT,
          const float* __restrict__ bkv, const float* __restrict__ ropek,
          _Float16* __restrict__ Kh, _Float16* __restrict__ Vt) {
  int lane = threadIdx.x & 31, wave = threadIdx.x >> 5;
  int m0 = blockIdx.y * 128 + wave * 32;  // over B*NKV = 16384
  int n0 = blockIdx.x * 64;               // over 1536
  v8f c[2][4] = {};
  gemm32x64(dat + (size_t)m0 * 768, WkvT + (size_t)n0 * 768, 768, lane, c);
  int colb = lane & 15;
  int rowOff = (lane < 16) ? 0 : 8;
#pragma unroll
  for (int mh = 0; mh < 2; ++mh)
#pragma unroll
    for (int f = 0; f < 4; ++f) {
      float bb = bkv[n0 + f * 16 + colb];
#pragma unroll
      for (int r = 0; r < 8; ++r) c[mh][f][r] += bb;
    }
  if (n0 < 768) {  // K half: apply RoPE
    int hh = n0 >> 6;
#pragma unroll
    for (int mh = 0; mh < 2; ++mh)
#pragma unroll
      for (int r = 0; r < 8; ++r) {
        int row = m0 + mh * 16 + rowOff + r;
        int n = row & 4095, b = row >> 12;
        const float* rk = ropek + n * 128;
        float s0 = rk[colb],      s1 = rk[16 + colb], s2 = rk[32 + colb], s3 = rk[48 + colb];
        float c0 = rk[64 + colb], c1 = rk[80 + colb], c2 = rk[96 + colb], c3 = rk[112 + colb];
        float x0 = c[mh][0][r], x1 = c[mh][1][r], x2 = c[mh][2][r], x3 = c[mh][3][r];
        _Float16* o = Kh + (((size_t)(b * 12 + hh) * 4096 + n) * 64);
        o[colb]      = (_Float16)(x0 * c0 - x2 * s0);
        o[16 + colb] = (_Float16)(x1 * c1 - x3 * s1);
        o[32 + colb] = (_Float16)(x2 * c2 + x0 * s2);
        o[48 + colb] = (_Float16)(x3 * c3 + x1 * s3);
      }
  } else {  // V half: store d-major (transposed)
    int hh = (n0 - 768) >> 6;
#pragma unroll
    for (int mh = 0; mh < 2; ++mh)
#pragma unroll
      for (int r = 0; r < 8; ++r) {
        int row = m0 + mh * 16 + rowOff + r;
        int n = row & 4095, b = row >> 12;
        _Float16* vb = Vt + ((size_t)(b * 12 + hh) * 64) * 4096;
#pragma unroll
        for (int f = 0; f < 4; ++f)
          vb[(size_t)(f * 16 + colb) * 4096 + n] = (_Float16)c[mh][f][r];
      }
  }
}

// ------- flash attention: 4 waves/block share LDS-staged K/V; 1 q-tile per wave -------
__global__ void __launch_bounds__(128)
k_attn(const _Float16* __restrict__ Qh, const _Float16* __restrict__ Kh,
       const _Float16* __restrict__ Vt, _Float16* __restrict__ Oh) {
  int tid = threadIdx.x;
  int lane = tid & 31, wave = tid >> 5;
  int qt = blockIdx.x * 4 + wave;
  int h = blockIdx.y, b = blockIdx.z;
  size_t bh = (size_t)b * 12 + h;
  const _Float16* Qb = Qh + (bh * 512 + (size_t)qt * 16) * 64;
  const _Float16* Kb = Kh + bh * 4096 * 64;
  const _Float16* Vb = Vt + bh * 64 * 4096;
  int colb = lane & 15;
  int rowOff = (lane < 16) ? 0 : 8;

  __shared__ __align__(32) _Float16 Ktile[32 * 64];   // [key][d]
  __shared__ __align__(32) _Float16 Vtile[64 * 32];   // [d][key]
  __shared__ __align__(32) _Float16 plds[4][16 * 32];

  v16h qa[2];
#pragma unroll
  for (int kc = 0; kc < 2; ++kc) qa[kc] = ldA(Qb + kc * 32, 64, lane);

  v8f o[4] = {};
  float mrow[8], lsum[8];
#pragma unroll
  for (int r = 0; r < 8; ++r) { mrow[r] = -3.0e38f; lsum[r] = 0.0f; }

  // staging indices (per thread: 16 halfs of K, 16 halfs of V)
  int koff = tid * 16;            // 0..2047 over Ktile
  int krow = koff >> 6, kd = koff & 63;
  int vd = tid >> 1, vk = (tid & 1) * 16;

  for (int kb = 0; kb < 4096; kb += 32) {
    __syncthreads();  // previous iteration's consumers done
#if HAVE_ASYNC_LDS
    {
      const _Float16* gk = Kb + (size_t)(kb + krow) * 64 + kd;
      const _Float16* gv = Vb + (size_t)vd * 4096 + kb + vk;
      __builtin_amdgcn_global_load_async_to_lds_b128((g_v4i*)gk,       (l_v4i*)&Ktile[koff],           0, 0);
      __builtin_amdgcn_global_load_async_to_lds_b128((g_v4i*)(gk + 8), (l_v4i*)&Ktile[koff + 8],       0, 0);
      __builtin_amdgcn_global_load_async_to_lds_b128((g_v4i*)gv,       (l_v4i*)&Vtile[vd * 32 + vk],     0, 0);
      __builtin_amdgcn_global_load_async_to_lds_b128((g_v4i*)(gv + 8), (l_v4i*)&Vtile[vd * 32 + vk + 8], 0, 0);
      __builtin_amdgcn_s_wait_asynccnt(0);
    }
#else
    *(v16h*)&Ktile[koff]         = *(const v16h*)(Kb + (size_t)(kb + krow) * 64 + kd);
    *(v16h*)&Vtile[vd * 32 + vk] = *(const v16h*)(Vb + (size_t)vd * 4096 + kb + vk);
#endif
    __syncthreads();

    // S(16x32) = Q(16x64) * K^T  (two 16-key tiles, two d-halves each) from LDS
    v8f s[2] = {};
#pragma unroll
    for (int t = 0; t < 2; ++t)
#pragma unroll
      for (int kc = 0; kc < 2; ++kc) {
        v16h kf = ldB(Ktile + (t * 16) * 64 + kc * 32, 64, lane);
        s[t] = WMMA_F16(qa[kc], kf, s[t]);
      }
    // online softmax (rows live across 16-lane halves)
    float p0[8], p1[8];
#pragma unroll
    for (int r = 0; r < 8; ++r) {
      float t = fmaxf(s[0][r], s[1][r]);
      t = fmaxf(t, __shfl_xor(t, 1, 32));
      t = fmaxf(t, __shfl_xor(t, 2, 32));
      t = fmaxf(t, __shfl_xor(t, 4, 32));
      t = fmaxf(t, __shfl_xor(t, 8, 32));
      float mn = fmaxf(mrow[r], t);
      float alpha = __expf(mrow[r] - mn);
      mrow[r] = mn;
      float e0 = __expf(s[0][r] - mn);
      float e1 = __expf(s[1][r] - mn);
      p0[r] = e0; p1[r] = e1;
      float rs = e0 + e1;
      rs += __shfl_xor(rs, 1, 32);
      rs += __shfl_xor(rs, 2, 32);
      rs += __shfl_xor(rs, 4, 32);
      rs += __shfl_xor(rs, 8, 32);
      lsum[r] = lsum[r] * alpha + rs;
      o[0][r] *= alpha; o[1][r] *= alpha; o[2][r] *= alpha; o[3][r] *= alpha;
    }
    // C-layout P -> per-wave LDS buffer -> A-layout P (16x32); wave-internal, DS is in-order
    _Float16* pw = plds[wave];
#pragma unroll
    for (int r = 0; r < 8; ++r) {
      pw[(rowOff + r) * 32 + colb]      = (_Float16)p0[r];
      pw[(rowOff + r) * 32 + 16 + colb] = (_Float16)p1[r];
    }
    __syncthreads();
    v16h pa = ldA(pw, 32, lane);
    // O(16x64) += P(16x32) * V(32x64); Vtile is d-major so B-frags are contiguous
#pragma unroll
    for (int f = 0; f < 4; ++f) {
      v16h vf = ldB(Vtile + (f * 16) * 32, 32, lane);
      o[f] = WMMA_F16(pa, vf, o[f]);
    }
  }
  // normalize + store Oh [B][NQ][768]
#pragma unroll
  for (int r = 0; r < 8; ++r) {
    float inv = 1.0f / lsum[r];
    int q = qt * 16 + rowOff + r;
    _Float16* out = Oh + ((size_t)b * 512 + q) * 768 + h * 64;
    out[colb]      = (_Float16)(o[0][r] * inv);
    out[16 + colb] = (_Float16)(o[1][r] * inv);
    out[32 + colb] = (_Float16)(o[2][r] * inv);
    out[48 + colb] = (_Float16)(o[3][r] * inv);
  }
}

// ---------------- output projection -> f32 d_out ----------------
__global__ void __launch_bounds__(128)
k_gemm_proj(const _Float16* __restrict__ Oh, const _Float16* __restrict__ WpT,
            const float* __restrict__ bp, float* __restrict__ out) {
  int lane = threadIdx.x & 31, wave = threadIdx.x >> 5;
  int m0 = blockIdx.y * 128 + wave * 32;
  int n0 = blockIdx.x * 64;
  v8f c[2][4] = {};
  gemm32x64(Oh + (size_t)m0 * 768, WpT + (size_t)n0 * 768, 768, lane, c);
  int colb = lane & 15;
  int rowOff = (lane < 16) ? 0 : 8;
#pragma unroll
  for (int mh = 0; mh < 2; ++mh)
#pragma unroll
    for (int f = 0; f < 4; ++f) {
      float bb = bp[n0 + f * 16 + colb];
#pragma unroll
      for (int r = 0; r < 8; ++r)
        out[(size_t)(m0 + mh * 16 + rowOff + r) * 768 + n0 + f * 16 + colb] = c[mh][f][r] + bb;
    }
}

extern "C" void kernel_launch(void* const* d_in, const int* in_sizes, int n_in,
                              void* d_out, int out_size, void* d_ws, size_t ws_size,
                              hipStream_t stream) {
  (void)in_sizes; (void)n_in; (void)out_size; (void)ws_size;
  const float* latent = (const float*)d_in[0];
  const float* data   = (const float*)d_in[1];
  const float* ropeq  = (const float*)d_in[2];
  const float* ropek  = (const float*)d_in[3];
  const float* Wq     = (const float*)d_in[4];
  const float* bq     = (const float*)d_in[5];
  const float* Wkv    = (const float*)d_in[6];
  const float* bkv    = (const float*)d_in[7];
  const float* Wproj  = (const float*)d_in[8];
  const float* bproj  = (const float*)d_in[9];
  float* out = (float*)d_out;

  char* ws = (char*)d_ws;
  size_t off = 0;
  auto alloc = [&](size_t bytes) -> void* {
    void* p = ws + off;
    off += (bytes + 255) & ~(size_t)255;
    return p;
  };
  _Float16* lat_h = (_Float16*)alloc((size_t)2048 * 768 * 2);
  _Float16* dat_h = (_Float16*)alloc((size_t)16384 * 768 * 2);
  _Float16* WqT   = (_Float16*)alloc((size_t)768 * 768 * 2);
  _Float16* WkvT  = (_Float16*)alloc((size_t)1536 * 768 * 2);
  _Float16* WpT   = (_Float16*)alloc((size_t)768 * 768 * 2);
  _Float16* QhB   = (_Float16*)alloc((size_t)4 * 12 * 512 * 64 * 2);
  _Float16* KhB   = (_Float16*)alloc((size_t)4 * 12 * 4096 * 64 * 2);
  _Float16* VtB   = (_Float16*)alloc((size_t)4 * 12 * 64 * 4096 * 2);
  _Float16* OhB   = (_Float16*)alloc((size_t)2048 * 768 * 2);

  k_f32_to_f16<<<(2048 * 768 + 255) / 256, 256, 0, stream>>>(latent, lat_h, 2048 * 768);
  k_f32_to_f16<<<(16384 * 768 + 255) / 256, 256, 0, stream>>>(data, dat_h, 16384 * 768);
  k_transpose_f16<<<(768 * 768 + 255) / 256, 256, 0, stream>>>(Wq, WqT, 768, 768);
  k_transpose_f16<<<(768 * 1536 + 255) / 256, 256, 0, stream>>>(Wkv, WkvT, 768, 1536);
  k_transpose_f16<<<(768 * 768 + 255) / 256, 256, 0, stream>>>(Wproj, WpT, 768, 768);

  k_gemm_q<<<dim3(12, 16), 128, 0, stream>>>(lat_h, WqT, bq, ropeq, QhB);
  k_gemm_kv<<<dim3(24, 128), 128, 0, stream>>>(dat_h, WkvT, bkv, ropek, KhB, VtB);
  k_attn<<<dim3(8, 12, 4), 128, 0, stream>>>(QhB, KhB, VtB, OhB);
  k_gemm_proj<<<dim3(12, 16), 128, 0, stream>>>(OhB, WpT, bproj, out);
}
